// Hgru3_45775761441122
// MI455X (gfx1250) — matrix operand order repaired and checked by reading, not compiled, not run on hardware
//
#include <hip/hip_runtime.h>
#include <hip/hip_bf16.h>

// ---------------------------------------------------------------------------
// Problem constants (match reference: B=2, N=8192, E=1024, D=128, H=8, CHUNK=64)
// ---------------------------------------------------------------------------
constexpr int Bc = 2, Nc = 8192, Ec = 1024, Dc = 128, Hc = 8, CHc = 64;
constexpr int GLA_LD  = 136;  // stride (halves) for 128-wide row-major tiles (mult of 8)
constexpr int GLA_LDT = 72;   // stride (halves) for 64-wide transposed tiles (mult of 4)
constexpr int GLA_LDA = 72;   // stride (halves) for 64-wide A tile          (mult of 4)

typedef __attribute__((ext_vector_type(8)))  float  v8f;
typedef __attribute__((ext_vector_type(16))) __bf16 v16bf;

union BF16x16 {
    v16bf v;
    unsigned int u[8];
    __bf16 h[16];
};
union BF16x8 {
    uint4 u;
    __bf16 h[8];
};

__device__ inline v8f wmma_bf16(v16bf a, v16bf b, v8f c) {
    return __builtin_amdgcn_wmma_f32_16x16x32_bf16(false, a, false, b, (short)0, c,
                                                   false, false);
}

// A operand (16x32 bf16) from row-major LDS matrix: A[m][kk] = base[(row0+m)*ld + col0+kk]
// Packed dword pairs (kk even, ld multiple of 4).
__device__ inline v16bf load_wmma_A(const __bf16* base, int row0, int col0, int ld) {
    int l  = threadIdx.x & 31;
    int m  = row0 + (l & 15);
    int hh = l >> 4;
    const __bf16* p = base + m * ld + col0;
    BF16x16 r;
#pragma unroll
    for (int i = 0; i < 8; ++i) {
        int kk = ((i >> 2) << 4) + (hh << 3) + ((i & 3) << 1);  // even
        r.u[i] = *(const unsigned int*)(p + kk);
    }
    return r.v;
}

// B operand (32x16 bf16) with K contiguous: B[kk][n] = base[(n0+n)*ld + k0+kk]
// (i.e. the source holds the B matrix transposed, row-major). Packed dword pairs.
__device__ inline v16bf load_wmma_B_T(const __bf16* base, int k0, int n0, int ld) {
    int l  = threadIdx.x & 31;
    int n  = n0 + (l & 15);
    int hh = l >> 4;
    const __bf16* p = base + n * ld + k0;
    BF16x16 r;
#pragma unroll
    for (int i = 0; i < 8; ++i) {
        int kk = ((i >> 2) << 4) + (hh << 3) + ((i & 3) << 1);  // even
        r.u[i] = *(const unsigned int*)(p + kk);
    }
    return r.v;
}

__device__ inline float sigmoidf_(float x) { return 1.f / (1.f + expf(-x)); }
__device__ inline float siluf_(float x)    { return x * sigmoidf_(x); }

// ---------------------------------------------------------------------------
// f32 -> bf16 conversion
// ---------------------------------------------------------------------------
__global__ __launch_bounds__(256) void cvt_bf16_kernel(const float* __restrict__ in,
                                                       __bf16* __restrict__ out, size_t n) {
    size_t i = (size_t)blockIdx.x * 256 + threadIdx.x;
    if (i < n) out[i] = (__bf16)in[i];
}

// ---------------------------------------------------------------------------
// Tiled bf16 WMMA GEMM: C[M,N] = act(A[M,K] @ B[K,N]); A,B bf16 row-major.
// Block tile 128x128 (or 128xN if N<128), 8 waves, each wave 32x64 (2x4 tiles).
// ACT: 0 = none, 1 = silu.  CT: float or __bf16.
// ---------------------------------------------------------------------------
template <int ACT, typename CT>
__global__ __launch_bounds__(256) void gemm_bf16_kernel(const __bf16* __restrict__ A,
                                                        const __bf16* __restrict__ B,
                                                        CT* __restrict__ C,
                                                        int M, int N, int K) {
    __shared__ __bf16 sA[128 * 36];   // [row][k], stride 36
    __shared__ __bf16 sB[128 * 36];   // transposed: [n][k], stride 36
    const int tid  = threadIdx.x;
    const int wave = tid >> 5;
    const int bm   = blockIdx.x * 128;
    const int bn   = blockIdx.y * 128;
    const int wm   = (wave >> 1) * 32;
    const int wn   = (wave & 1) * 64;
    const int BNW  = (N < 128) ? N : 128;     // valid block width (128 or 128-col pad)

    v8f acc[2][4] = {};

    for (int k0 = 0; k0 < K; k0 += 32) {
        // stage A: 128 rows x 32 halves = 512 uint4
        for (int t = tid; t < 512; t += 256) {
            int row = t >> 2, c8 = (t & 3) << 3;
            *(uint4*)(sA + row * 36 + c8) =
                *(const uint4*)(A + (size_t)(bm + row) * K + k0 + c8);
        }
        // stage B transposed: 32 rows x BNW cols; uint4 global reads, b16 scatter
        for (int t = tid; t < 4 * BNW / 8 * 8; t += 256) {      // 32*BNW/8 uint4 groups
            int kk = t / (BNW >> 3), g8 = (t % (BNW >> 3)) << 3;
            BF16x8 tb;
            tb.u = *(const uint4*)(B + (size_t)(k0 + kk) * N + bn + g8);
#pragma unroll
            for (int u2 = 0; u2 < 8; ++u2) sB[(g8 + u2) * 36 + kk] = tb.h[u2];
        }
        if (k0 + 32 < K)
            __builtin_prefetch(A + (size_t)(bm + (tid >> 1)) * K + k0 + 32, 0, 1);
        __syncthreads();

        v16bf a0 = load_wmma_A(sA, wm, 0, 36);
        v16bf a1 = load_wmma_A(sA, wm + 16, 0, 36);
#pragma unroll
        for (int j = 0; j < 4; ++j) {
            int nj = wn + j * 16;
            if (nj < BNW) {
                v16bf bb = load_wmma_B_T(sB, 0, nj, 36);
                acc[0][j] = wmma_bf16(a0, bb, acc[0][j]);
                acc[1][j] = wmma_bf16(a1, bb, acc[1][j]);
            }
        }
        __syncthreads();
    }

    const int l  = tid & 31;
    const int cn = l & 15;
    const int rh = l >> 4;
#pragma unroll
    for (int mt = 0; mt < 2; ++mt)
#pragma unroll
        for (int nt = 0; nt < 4; ++nt) {
            int n = bn + wn + nt * 16 + cn;
            if (wn + nt * 16 >= BNW) continue;
#pragma unroll
            for (int i = 0; i < 8; ++i) {
                int m = bm + wm + mt * 16 + rh * 8 + i;
                float v = acc[mt][nt][i];
                if (ACT == 1) v = siluf_(v);
                C[(size_t)m * N + n] = (CT)v;
            }
        }
}

// ---------------------------------------------------------------------------
// f_pre = x@Wf  ->  log_f[b,h,n] = log(lb + (1-lb)*sigmoid(f_pre))
// ---------------------------------------------------------------------------
__global__ __launch_bounds__(256) void logf_kernel(const float* __restrict__ x,
                                                   const float* __restrict__ Wf,
                                                   const float* __restrict__ log_lb,
                                                   float* __restrict__ log_f) {
    int idx = blockIdx.x * 256 + threadIdx.x;     // over B*N*H
    int row = idx >> 3;                            // b*N + n
    int h   = idx & 7;
    float acc = 0.f;
    const float* xr = x + (size_t)row * Ec;
    for (int e = 0; e < Ec; ++e) acc += xr[e] * Wf[(size_t)e * Hc + h];
    float lb = expf(log_lb[h]);
    float f  = lb + (1.f - lb) * sigmoidf_(acc);
    int b = row / Nc, n = row - b * Nc;
    log_f[((size_t)(b * Hc + h)) * Nc + n] = logf(f);
}

// ---------------------------------------------------------------------------
// Householder reflection of q (bf16 in/out), beta from b1@Wb2 per row.
// ---------------------------------------------------------------------------
__global__ __launch_bounds__(256) void reflect_kernel(__bf16* __restrict__ q,
                                                      const float* __restrict__ b1,
                                                      const float* __restrict__ Wb2) {
    __shared__ float sB1[Dc];
    __shared__ float sBeta[Ec];
    __shared__ float red[256];
    const int row = blockIdx.x;
    const int tid = threadIdx.x;
    if (tid < Dc) sB1[tid] = b1[(size_t)row * Dc + tid];
    __syncthreads();
    for (int c = tid; c < Ec; c += 256) {
        float a = 0.f;
        for (int i = 0; i < Dc; ++i) a += sB1[i] * Wb2[(size_t)i * Ec + c];
        sBeta[c] = siluf_(a);
    }
    __syncthreads();
    float p = 0.f;
    for (int c = tid; c < Ec; c += 256) p += sBeta[c] * sBeta[c];
    red[tid] = p;
    __syncthreads();
    for (int s = 128; s > 0; s >>= 1) {
        if (tid < s) red[tid] += red[tid + s];
        __syncthreads();
    }
    float nb   = sqrtf(red[0]);
    float invn = 1.f / (fmaxf(nb, 1e-6f) * 32.f);   // 32 = sqrt(E)
    __syncthreads();

    __bf16* qr = q + (size_t)row * Ec;
    float d = 0.f;
    for (int c = tid; c < Ec; c += 256) d += (float)qr[c] * sBeta[c];
    red[tid] = d;
    __syncthreads();
    for (int s = 128; s > 0; s >>= 1) {
        if (tid < s) red[tid] += red[tid + s];
        __syncthreads();
    }
    float coef = 2.f * red[0] * invn * invn;
    for (int c = tid; c < Ec; c += 256) {
        float qq = (float)qr[c];
        qr[c] = (__bf16)(qq - coef * sBeta[c]);
    }
}

// ---------------------------------------------------------------------------
// Chunked GLA scan. One workgroup (8 waves) per (b,h). All operand loads are
// packed-pair dword LDS reads; decay scalings folded into staging.
// ---------------------------------------------------------------------------
__global__ __launch_bounds__(256) void gla_kernel(const __bf16* __restrict__ q,
                                                  const __bf16* __restrict__ k,
                                                  const __bf16* __restrict__ v,
                                                  const float* __restrict__ log_f,
                                                  float* __restrict__ o) {
    extern __shared__ char smem[];
    __bf16* sQ   = (__bf16*)smem;              // 64  x GLA_LD   raw q       (intra A)
    __bf16* sQg  = sQ  + 64 * GLA_LD;          // 64  x GLA_LD   q*exp(g)    (o_inter A)
    __bf16* sK   = sQg + 64 * GLA_LD;          // 64  x GLA_LD   raw k       (intra B^T)
    __bf16* sKT  = sK  + 64 * GLA_LD;          // 128 x GLA_LDT  kd^T        (S-update A)
    __bf16* sVT  = sKT + 128 * GLA_LDT;        // 128 x GLA_LDT  v^T         (B for A@V, kd^T@V)
    __bf16* sST  = sVT + 128 * GLA_LDT;        // 128 x GLA_LD   S^T bf16    (o_inter B)
    __bf16* sA   = sST + 128 * GLA_LD;         // 64  x GLA_LDA  decayed A
    float*  sg   = (float*)(sA + 64 * GLA_LDA);  // 64 cum-log-decay
    float*  slog = sg + 64;                      // 64 raw log_f

    const int tid  = threadIdx.x;
    const int wave = tid >> 5;
    const int lane = tid & 31;
    const int bh   = blockIdx.x;
    const int b    = bh / Hc;
    const int h    = bh - b * Hc;
    const size_t headoff = (size_t)h * Dc;

    for (int i = tid; i < 128 * GLA_LD; i += 256) sST[i] = (__bf16)0.f;

    v8f Sacc[8] = {};                       // wave strip: S rows wave*16..+16, all 128 cols

    const int o_r0 = (wave >> 1) * 16;      // output rows (of 64)
    const int o_c0 = (wave & 1) * 64;       // output cols (of 128), 4 tiles
    const int s_m0 = wave * 16;             // state strip rows

    __syncthreads();

    for (int c = 0; c < Nc / CHc; ++c) {
        const int n0 = c * CHc;
        if (tid < 64) slog[tid] = log_f[((size_t)(b * Hc + h)) * Nc + n0 + tid];
        __syncthreads();
        if (tid == 0) {
            float r = 0.f;
            for (int i = 0; i < 64; ++i) { r += slog[i]; sg[i] = r; }
        }
        __syncthreads();
        const float gl = sg[63];

        // ---- stage: 64 rows x 16 uint4 groups (8 halves) per matrix ----
        for (int t = tid; t < 64 * 16; t += 256) {
            int r = t >> 4, g8 = (t & 15) << 3;
            size_t goff = ((size_t)(b * Nc + n0 + r)) * Ec + headoff + g8;
            const float eq = expf(sg[r]);
            const float ek = expf(gl - sg[r]);
            // q: raw + scaled copy
            BF16x8 tq;
            tq.u = *(const uint4*)(q + goff);
            *(uint4*)(sQ + r * GLA_LD + g8) = tq.u;
#pragma unroll
            for (int u2 = 0; u2 < 8; ++u2) tq.h[u2] = (__bf16)((float)tq.h[u2] * eq);
            *(uint4*)(sQg + r * GLA_LD + g8) = tq.u;
            // k: raw + decayed transposed
            BF16x8 tk;
            tk.u = *(const uint4*)(k + goff);
            *(uint4*)(sK + r * GLA_LD + g8) = tk.u;
#pragma unroll
            for (int u2 = 0; u2 < 8; ++u2)
                sKT[(g8 + u2) * GLA_LDT + r] = (__bf16)((float)tk.h[u2] * ek);
            // v: transposed
            BF16x8 tv;
            tv.u = *(const uint4*)(v + goff);
#pragma unroll
            for (int u2 = 0; u2 < 8; ++u2) sVT[(g8 + u2) * GLA_LDT + r] = tv.h[u2];
        }
        __syncthreads();

        // ---- intra: A = q @ k^T (raw), then decay+mask elementwise ----
        const int a_r0 = (wave >> 1) * 16;
        v8f Aacc[2];
#pragma unroll
        for (int j = 0; j < 2; ++j) {
            int a_c0 = ((wave & 1) * 2 + j) * 16;
            v8f acc = {};
#pragma unroll
            for (int kk = 0; kk < Dc; kk += 32) {
                v16bf a  = load_wmma_A(sQ, a_r0, kk, GLA_LD);
                v16bf bb = load_wmma_B_T(sK, kk, a_c0, GLA_LD);
                acc = wmma_bf16(a, bb, acc);
            }
            Aacc[j] = acc;
        }
        // decay + causal mask, store A as bf16 (region disjoint from sQ/sK)
#pragma unroll
        for (int j = 0; j < 2; ++j) {
            int a_c0 = ((wave & 1) * 2 + j) * 16;
#pragma unroll
            for (int i = 0; i < 8; ++i) {
                int mi = a_r0 + (lane >> 4) * 8 + i;
                int nj = a_c0 + (lane & 15);
                float av = Aacc[j][i];
                av = (nj <= mi) ? av * expf(sg[mi] - sg[nj]) : 0.f;
                sA[mi * GLA_LDA + nj] = (__bf16)av;
            }
        }
        __syncthreads();

        // ---- o = A @ V + qg @ S  (all B operands packed) ----
        v8f Oacc[4];
#pragma unroll
        for (int j = 0; j < 4; ++j) {
            v8f acc = {};
            int cj = o_c0 + j * 16;
#pragma unroll
            for (int kk = 0; kk < CHc; kk += 32) {
                v16bf a  = load_wmma_A(sA, o_r0, kk, GLA_LDA);
                v16bf bb = load_wmma_B_T(sVT, kk, cj, GLA_LDT);   // B[kk=i][n=d] = sVT[d][i]
                acc = wmma_bf16(a, bb, acc);
            }
#pragma unroll
            for (int kk = 0; kk < Dc; kk += 32) {
                v16bf a  = load_wmma_A(sQg, o_r0, kk, GLA_LD);
                v16bf bb = load_wmma_B_T(sST, kk, cj, GLA_LD);    // B[kk=d][n] = sST[n][d]
                acc = wmma_bf16(a, bb, acc);
            }
            Oacc[j] = acc;
        }
        __syncthreads();   // all sST reads done before overwrite

        // ---- state update: S = exp(gl)*S + kd^T @ V (register strips) ----
        const float egl = expf(gl);
#pragma unroll
        for (int ct = 0; ct < 8; ++ct) {
            v8f acc = Sacc[ct];
#pragma unroll
            for (int i = 0; i < 8; ++i) acc[i] = acc[i] * egl;
#pragma unroll
            for (int kk = 0; kk < CHc; kk += 32) {
                v16bf a  = load_wmma_A(sKT, s_m0, kk, GLA_LDT);   // A[m=d][kk=i] = sKT[d][i]
                v16bf bb = load_wmma_B_T(sVT, kk, ct * 16, GLA_LDT);
                acc = wmma_bf16(a, bb, acc);
            }
            Sacc[ct] = acc;
#pragma unroll
            for (int i = 0; i < 8; ++i) {
                int mi = s_m0 + (lane >> 4) * 8 + i;   // state row (d)
                int nj = ct * 16 + (lane & 15);        // state col (n)
                sST[nj * GLA_LD + mi] = (__bf16)acc[i];
            }
        }

        // ---- write o chunk ----
#pragma unroll
        for (int j = 0; j < 4; ++j)
#pragma unroll
            for (int i = 0; i < 8; ++i) {
                int mi = o_r0 + (lane >> 4) * 8 + i;
                int nj = o_c0 + j * 16 + (lane & 15);
                o[((size_t)(b * Nc + n0 + mi)) * Ec + headoff + nj] = Oacc[j][i];
            }
        __syncthreads();
    }
}

// ---------------------------------------------------------------------------
// gate = sigmoid(g1@Wg2); o = LN(o*gate)*scale + bias  -> bf16
// ---------------------------------------------------------------------------
__global__ __launch_bounds__(256) void gate_ln_kernel(const float* __restrict__ o,
                                                      const float* __restrict__ g1,
                                                      const float* __restrict__ Wg2,
                                                      const float* __restrict__ ln_s,
                                                      const float* __restrict__ ln_b,
                                                      __bf16* __restrict__ obf) {
    __shared__ float sG1[Dc];
    __shared__ float sRow[Ec];
    __shared__ float red[256];
    __shared__ float red2[256];
    const int row = blockIdx.x;
    const int tid = threadIdx.x;
    if (tid < Dc) sG1[tid] = g1[(size_t)row * Dc + tid];
    __syncthreads();
    const float* orow = o + (size_t)row * Ec;
    for (int c = tid; c < Ec; c += 256) {
        float a = 0.f;
        for (int i = 0; i < Dc; ++i) a += sG1[i] * Wg2[(size_t)i * Ec + c];
        sRow[c] = orow[c] * sigmoidf_(a);
    }
    __syncthreads();
    float s = 0.f, s2 = 0.f;
    for (int c = tid; c < Ec; c += 256) { float x = sRow[c]; s += x; s2 += x * x; }
    red[tid] = s; red2[tid] = s2;
    __syncthreads();
    for (int st = 128; st > 0; st >>= 1) {
        if (tid < st) { red[tid] += red[tid + st]; red2[tid] += red2[tid + st]; }
        __syncthreads();
    }
    float mean = red[0] * (1.f / Ec);
    float var  = red2[0] * (1.f / Ec) - mean * mean;
    float inv  = rsqrtf(var + 1e-5f);
    for (int c = tid; c < Ec; c += 256)
        obf[(size_t)row * Ec + c] = (__bf16)((sRow[c] - mean) * inv * ln_s[c] + ln_b[c]);
}

// ---------------------------------------------------------------------------
// Host launch
// ---------------------------------------------------------------------------
extern "C" void kernel_launch(void* const* d_in, const int* in_sizes, int n_in,
                              void* d_out, int out_size, void* d_ws, size_t ws_size,
                              hipStream_t stream) {
    (void)in_sizes; (void)n_in; (void)out_size; (void)ws_size;
    const float* x   = (const float*)d_in[0];
    const float* llb = (const float*)d_in[1];
    const float* Wq  = (const float*)d_in[2];
    const float* Wk  = (const float*)d_in[3];
    const float* Wv  = (const float*)d_in[4];
    const float* Wf  = (const float*)d_in[5];
    const float* Wb1 = (const float*)d_in[6];
    const float* Wb2 = (const float*)d_in[7];
    const float* Wg1 = (const float*)d_in[8];
    const float* Wg2 = (const float*)d_in[9];
    const float* Wo  = (const float*)d_in[10];
    const float* lns = (const float*)d_in[11];
    const float* lnb = (const float*)d_in[12];
    float* out = (float*)d_out;

    const size_t BN = (size_t)Bc * Nc;  // 16384
    char* ws = (char*)d_ws;
    size_t cur = 0;
    auto alloc = [&](size_t bytes) -> void* {
        void* p = ws + cur;
        cur = (cur + bytes + 255) & ~(size_t)255;
        return p;
    };

    __bf16* xbf   = (__bf16*)alloc(BN * Ec * 2);
    __bf16* Wqbf  = (__bf16*)alloc((size_t)Ec * Ec * 2);
    __bf16* Wkbf  = (__bf16*)alloc((size_t)Ec * Ec * 2);
    __bf16* Wvbf  = (__bf16*)alloc((size_t)Ec * Ec * 2);
    __bf16* Wobf  = (__bf16*)alloc((size_t)Ec * Ec * 2);
    __bf16* Wb1bf = (__bf16*)alloc((size_t)Ec * Dc * 2);
    __bf16* Wg1bf = (__bf16*)alloc((size_t)Ec * Dc * 2);
    __bf16* qbf   = (__bf16*)alloc(BN * Ec * 2);
    __bf16* kbf   = (__bf16*)alloc(BN * Ec * 2);
    __bf16* vbf   = (__bf16*)alloc(BN * Ec * 2);
    float*  b1    = (float*)alloc(BN * Dc * 4);
    float*  g1    = (float*)alloc(BN * Dc * 4);
    float*  lgf   = (float*)alloc((size_t)Bc * Hc * Nc * 4);
    float*  oatt  = (float*)alloc(BN * Ec * 4);
    __bf16* obf   = (__bf16*)alloc(BN * Ec * 2);

    auto cvt = [&](const float* src, __bf16* dst, size_t n) {
        int blocks = (int)((n + 255) / 256);
        cvt_bf16_kernel<<<blocks, 256, 0, stream>>>(src, dst, n);
    };
    cvt(x,   xbf,   BN * Ec);
    cvt(Wq,  Wqbf,  (size_t)Ec * Ec);
    cvt(Wk,  Wkbf,  (size_t)Ec * Ec);
    cvt(Wv,  Wvbf,  (size_t)Ec * Ec);
    cvt(Wo,  Wobf,  (size_t)Ec * Ec);
    cvt(Wb1, Wb1bf, (size_t)Ec * Dc);
    cvt(Wg1, Wg1bf, (size_t)Ec * Dc);

    dim3 gFull((unsigned)(BN / 128), Ec / 128);
    dim3 gNarrow((unsigned)(BN / 128), 1);
    gemm_bf16_kernel<1, __bf16><<<gFull, 256, 0, stream>>>(xbf, Wqbf, qbf, (int)BN, Ec, Ec);
    gemm_bf16_kernel<1, __bf16><<<gFull, 256, 0, stream>>>(xbf, Wkbf, kbf, (int)BN, Ec, Ec);
    gemm_bf16_kernel<0, __bf16><<<gFull, 256, 0, stream>>>(xbf, Wvbf, vbf, (int)BN, Ec, Ec);
    gemm_bf16_kernel<0, float><<<gNarrow, 256, 0, stream>>>(xbf, Wb1bf, b1, (int)BN, Dc, Ec);
    gemm_bf16_kernel<0, float><<<gNarrow, 256, 0, stream>>>(xbf, Wg1bf, g1, (int)BN, Dc, Ec);

    logf_kernel<<<(unsigned)(BN * Hc / 256), 256, 0, stream>>>(x, Wf, llb, lgf);
    reflect_kernel<<<(unsigned)BN, 256, 0, stream>>>(qbf, b1, Wb2);

    size_t glashm = ((size_t)3 * 64 * GLA_LD + 2 * 128 * GLA_LDT + 128 * GLA_LD
                     + 64 * GLA_LDA) * sizeof(__bf16) + 128 * sizeof(float);
    gla_kernel<<<Bc * Hc, 256, glashm, stream>>>(qbf, kbf, vbf, lgf, oatt);

    gate_ln_kernel<<<(unsigned)BN, 256, 0, stream>>>(oatt, g1, Wg2, lns, lnb, obf);

    gemm_bf16_kernel<0, float><<<gFull, 256, 0, stream>>>(obf, Wobf, out, (int)BN, Ec, Ec);
}
